// ORViT_81982335746434
// MI455X (gfx1250) — compile-verified, matrix operand order checked by reading, not run on hardware
//
#include <hip/hip_runtime.h>
#include <hip/hip_bf16.h>

typedef unsigned short u16;
typedef __bf16 v16bf __attribute__((ext_vector_type(16)));
typedef float  v8f   __attribute__((ext_vector_type(8)));

union Frag {
  v16bf bf;
  uint4 q[2];
};

__device__ __forceinline__ u16 f2bf(float f) {
  unsigned int u = __float_as_uint(f);
  unsigned int r = (u + 0x7FFFu + ((u >> 16) & 1u)) >> 16;
  return (u16)r;
}
__device__ __forceinline__ float bf2f(u16 h) {
  return __uint_as_float(((unsigned int)h) << 16);
}

// A-fragment (16x32 bf16, row-major source, ld in elements).
// ISA 7.12.2: lane L holds row (m0 + L%16); VGPR0..3 = K pairs at kg..kg+7,
// VGPR4..7 = K pairs at 16+kg..16+kg+7, kg = (L/16)*8.  Two 16B loads.
__device__ __forceinline__ void load_frag_rm(Frag& f, const u16* base, int ld,
                                             int row, int k0, int lane) {
  int kg = (lane >> 4) * 8;
  const u16* p = base + (size_t)row * ld + k0 + kg;
  f.q[0] = *(const uint4*)(p);
  f.q[1] = *(const uint4*)(p + 16);
}

// B-fragment from pre-packed weights: 32 contiguous bytes per lane.
__device__ __forceinline__ void load_frag_pk(Frag& f, const u16* Bp, int kt,
                                             int nt, int NT, int lane) {
  const u16* p = Bp + (((size_t)kt * NT + nt) * 32 + lane) * 16;
  f.q[0] = ((const uint4*)p)[0];
  f.q[1] = ((const uint4*)p)[1];
}

__device__ __forceinline__ v8f wmma_bf16(const Frag& a, const Frag& b, v8f c) {
  return __builtin_amdgcn_wmma_f32_16x16x32_bf16(false, a.bf, false, b.bf,
                                                 (short)0, c, false, false);
}

// ---------------- constants -------------------------------------------------
// B=2 T=8 HG=WG=14 O=6 D=768 HEADS=12 CROP=4
// HW=196  P=HW+O=202  S=F*P=1616  N'=1+S=1617  Nx=1+T*HW=1569  PPAD=224

// ---------------- zero fill -------------------------------------------------
__global__ __launch_bounds__(256) void zero_u16_kernel(u16* p, int n) {
  for (int i = blockIdx.x * 256 + threadIdx.x; i < n; i += gridDim.x * 256)
    p[i] = 0;
}

// ---------------- weight packing (f32 KxN -> bf16 fragment order) -----------
__global__ __launch_bounds__(256) void pack_w(const float* W, u16* out, int K,
                                              int N, int ld) {
  int NT = N / 16;
  int total = K * N;
  for (int idx = blockIdx.x * 256 + threadIdx.x; idx < total;
       idx += gridDim.x * 256) {
    int j = idx & 15;
    int lane = (idx >> 4) & 31;
    int tile = idx >> 9;
    int nt = tile % NT;
    int kt = tile / NT;
    int rr = j >> 1, c = j & 1;
    int kg = (lane >> 4) * 8;
    int k = kt * 32 +
            ((rr < 4) ? (kg + 2 * rr + c) : (16 + kg + 2 * (rr - 4) + c));
    int n = nt * 16 + (lane & 15);
    out[idx] = f2bf(W[(size_t)k * ld + n]);
  }
}

// ---------------- ROI crop + MLP + maxpool (WMMA) ---------------------------
__global__ __launch_bounds__(256) void roi_obj_kernel(
    const float* __restrict__ x, const float* __restrict__ md,
    const u16* __restrict__ Wp1p, const u16* __restrict__ Wp2p,
    float* __restrict__ obj) {
  __shared__ u16 crop[16 * 768];
  __shared__ u16 hid[16 * 384];
  int id = blockIdx.x;               // (b*8+f)*6+o
  int o = id % 6, f = (id / 6) % 8, b = id / 48;
  int t = threadIdx.x, lane = t & 31, wid = t >> 5;

  const float* bx = md + (((size_t)(b * 6 + o)) * 8 + f) * 4;
  const float sc = 14.0f / 224.0f;
  float x1 = bx[0] * sc, y1 = bx[1] * sc, x2 = bx[2] * sc, y2 = bx[3] * sc;
  const float* fb = x + ((size_t)b * 1569 + 1 + f * 196) * 768;

  for (int cp = 0; cp < 16; ++cp) {
    int iy = cp >> 2, ix = cp & 3;
    float xs = x1 + (ix + 0.5f) * (x2 - x1) * 0.25f - 0.5f;
    float ys = y1 + (iy + 0.5f) * (y2 - y1) * 0.25f - 0.5f;
    xs = fminf(fmaxf(xs, 0.0f), 13.0f);
    ys = fminf(fmaxf(ys, 0.0f), 13.0f);
    float x0f = floorf(xs), y0f = floorf(ys);
    float wxv = xs - x0f, wyv = ys - y0f;
    int x0 = (int)x0f, y0 = (int)y0f;
    int x1i = min(x0 + 1, 13), y1i = min(y0 + 1, 13);
    const float* p00 = fb + (size_t)(y0 * 14 + x0) * 768;
    const float* p01 = fb + (size_t)(y0 * 14 + x1i) * 768;
    const float* p10 = fb + (size_t)(y1i * 14 + x0) * 768;
    const float* p11 = fb + (size_t)(y1i * 14 + x1i) * 768;
    float w00 = (1 - wyv) * (1 - wxv), w01 = (1 - wyv) * wxv;
    float w10 = wyv * (1 - wxv), w11 = wyv * wxv;
    for (int d = t; d < 768; d += 256)
      crop[cp * 768 + d] =
          f2bf(w00 * p00[d] + w01 * p01[d] + w10 * p10[d] + w11 * p11[d]);
  }
  __syncthreads();

  // hidden = relu(crop @ Wp1)  (16x768)@(768x384)
  for (int nt = wid; nt < 24; nt += 8) {
    v8f acc = {};
    for (int kt = 0; kt < 24; ++kt) {
      Frag a, bw;
      load_frag_rm(a, crop, 768, lane & 15, kt * 32, lane);
      load_frag_pk(bw, Wp1p, kt, nt, 24, lane);
      acc = wmma_bf16(a, bw, acc);
    }
    for (int r = 0; r < 8; ++r) {
      int m = r + 8 * (lane >> 4);
      hid[m * 384 + nt * 16 + (lane & 15)] = f2bf(fmaxf(acc[r], 0.0f));
    }
  }
  __syncthreads();

  // out = relu(hidden @ Wp2) then max over 16 crop points
  for (int nt = wid; nt < 48; nt += 8) {
    v8f acc = {};
    for (int kt = 0; kt < 12; ++kt) {
      Frag a, bw;
      load_frag_rm(a, hid, 384, lane & 15, kt * 32, lane);
      load_frag_pk(bw, Wp2p, kt, nt, 48, lane);
      acc = wmma_bf16(a, bw, acc);
    }
    float mx = acc[0];
    for (int r = 1; r < 8; ++r) mx = fmaxf(mx, acc[r]);
    mx = fmaxf(mx, 0.0f);  // relu then max == max(0, max)
    float other = __shfl_xor(mx, 16, 32);
    mx = fmaxf(mx, other);
    if (lane < 16)
      obj[(((size_t)(b * 8 + f)) * 6 + o) * 768 + nt * 16 + lane] = mx;
  }
}

// ---------------- layernorm over concat(cls, feat, obj) -> bf16 -------------
__global__ __launch_bounds__(256) void ln_kernel(const float* __restrict__ x,
                                                 const float* __restrict__ obj,
                                                 const float* __restrict__ g,
                                                 const float* __restrict__ bia,
                                                 u16* __restrict__ lnb) {
  __shared__ float red[256];
  __shared__ float red2[256];
  int rowg = blockIdx.x;
  int b = rowg / 1617, tok = rowg % 1617;
  const float* src;
  if (tok == 0) {
    src = x + (size_t)b * 1569 * 768;
  } else {
    int tt = tok - 1, f = tt / 202, r = tt % 202;
    if (r < 196)
      src = x + ((size_t)b * 1569 + 1 + f * 196 + r) * 768;
    else
      src = obj + (((size_t)(b * 8 + f)) * 6 + (r - 196)) * 768;
  }
  int t = threadIdx.x;
  float s = 0.0f, s2 = 0.0f;
  for (int d = t; d < 768; d += 256) {
    float v = src[d];
    s += v;
    s2 += v * v;
  }
  red[t] = s;
  red2[t] = s2;
  __syncthreads();
  for (int ofs = 128; ofs; ofs >>= 1) {
    if (t < ofs) {
      red[t] += red[t + ofs];
      red2[t] += red2[t + ofs];
    }
    __syncthreads();
  }
  float mu = red[0] * (1.0f / 768.0f);
  float var = red2[0] * (1.0f / 768.0f) - mu * mu;
  float rs = rsqrtf(var + 1e-5f);
  for (int d = t; d < 768; d += 256)
    lnb[(size_t)rowg * 768 + d] = f2bf((src[d] - mu) * rs * g[d] + bia[d]);
}

// ---------------- generic WMMA GEMM (A bf16 row-major, packed B) ------------
// Block = 8 waves in a 2(M) x 4(N) grid; each wave owns a 32x64 macro-tile
// (2 A-frags x 4 B-frags -> 8 WMMAs per K step) for register-level reuse.
// mode 0: QKV   -> qkvb bf16 (rows x 2304) + v transposed into vt2
// mode 1: bf16  -> outb bf16 (rows x 768), scaled
// mode 2: proj  -> d_out f32 with bias + residual + token re-slice
__global__ __launch_bounds__(256) void gemm_wmma(
    const u16* __restrict__ A, const u16* __restrict__ Bp, int rows, int K,
    int NT, int mode, float scale, u16* __restrict__ outb,
    float* __restrict__ outf, const float* __restrict__ bias,
    const float* __restrict__ xres, u16* __restrict__ qkvb,
    u16* __restrict__ vt2) {
  int lane = threadIdx.x & 31, wid = threadIdx.x >> 5;
  int wm = wid >> 2;                      // 0..1
  int wn = wid & 3;                       // 0..3
  int m0 = blockIdx.x * 64 + wm * 32;     // wave M base (2 x 16-row frags)
  int ntb = blockIdx.y * 16 + wn * 4;     // wave N base in 16-col tiles

  int row0 = m0 + (lane & 15);
  int row1 = m0 + 16 + (lane & 15);
  if (row0 >= rows) row0 = rows - 1;
  if (row1 >= rows) row1 = rows - 1;

  v8f acc[2][4];
  for (int mi = 0; mi < 2; ++mi)
    for (int ni = 0; ni < 4; ++ni) acc[mi][ni] = {};

  int ksteps = K >> 5;
  for (int kt = 0; kt < ksteps; ++kt) {
    Frag a0, a1, b0, b1, b2, b3;
    load_frag_rm(a0, A, K, row0, kt * 32, lane);
    load_frag_rm(a1, A, K, row1, kt * 32, lane);
    load_frag_pk(b0, Bp, kt, ntb + 0, NT, lane);
    load_frag_pk(b1, Bp, kt, ntb + 1, NT, lane);
    load_frag_pk(b2, Bp, kt, ntb + 2, NT, lane);
    load_frag_pk(b3, Bp, kt, ntb + 3, NT, lane);
    acc[0][0] = wmma_bf16(a0, b0, acc[0][0]);
    acc[0][1] = wmma_bf16(a0, b1, acc[0][1]);
    acc[0][2] = wmma_bf16(a0, b2, acc[0][2]);
    acc[0][3] = wmma_bf16(a0, b3, acc[0][3]);
    acc[1][0] = wmma_bf16(a1, b0, acc[1][0]);
    acc[1][1] = wmma_bf16(a1, b1, acc[1][1]);
    acc[1][2] = wmma_bf16(a1, b2, acc[1][2]);
    acc[1][3] = wmma_bf16(a1, b3, acc[1][3]);
  }

  for (int mi = 0; mi < 2; ++mi) {
    int mbase = m0 + mi * 16 + 8 * (lane >> 4);
    for (int ni = 0; ni < 4; ++ni) {
      int n = (ntb + ni) * 16 + (lane & 15);
      for (int r = 0; r < 8; ++r) {
        int m = mbase + r;
        if (m >= rows) continue;
        float v = acc[mi][ni][r] * scale;
        if (mode == 0) {
          qkvb[(size_t)m * 2304 + n] = f2bf(v);
          if (n >= 1536) {
            int b2i = m / 1617, tok = m % 1617;
            if (tok >= 1) {
              int dv = (n - 1536) & 63, hh = (n - 1536) >> 6;
              int s = tok - 1, fi = s / 202, p = s % 202;
              vt2[((((size_t)b2i * 12 + hh) * 64 + dv) * 8 + fi) * 224 + p] =
                  f2bf(v);
            }
          }
        } else if (mode == 1) {
          outb[(size_t)m * 768 + n] = f2bf(v);
        } else {
          int b2i = m / 1617, tr = m % 1617;
          int ot = -1;
          if (tr == 0) {
            ot = 0;
          } else {
            int fi = (tr - 1) / 202, r2 = (tr - 1) % 202;
            if (r2 < 196) ot = 1 + fi * 196 + r2;
          }
          if (ot >= 0) {
            size_t oi = ((size_t)b2i * 1569 + ot) * 768 + n;
            outf[oi] = xres[oi] + v + bias[n];
          }
        }
      }
    }
  }
}

// ---------------- cls-token attention (VALU) --------------------------------
__global__ __launch_bounds__(256) void cls_attn_kernel(
    const u16* __restrict__ qkv, u16* __restrict__ attin) {
  __shared__ float qv[64];
  __shared__ float pr[1617];
  __shared__ float red[256];
  int b = blockIdx.x / 12, h = blockIdx.x % 12;
  const u16* base = qkv + (size_t)b * 1617 * 2304;
  int t = threadIdx.x;
  if (t < 64) qv[t] = bf2f(base[h * 64 + t]);
  __syncthreads();
  float lmax = -1e30f;
  for (int key = t; key < 1617; key += 256) {
    const u16* kp = base + (size_t)key * 2304 + 768 + h * 64;
    float dot = 0.0f;
    for (int d = 0; d < 64; ++d) dot += qv[d] * bf2f(kp[d]);
    dot *= 0.125f;
    pr[key] = dot;
    lmax = fmaxf(lmax, dot);
  }
  red[t] = lmax;
  __syncthreads();
  for (int ofs = 128; ofs; ofs >>= 1) {
    if (t < ofs) red[t] = fmaxf(red[t], red[t + ofs]);
    __syncthreads();
  }
  float M = red[0];
  __syncthreads();
  float ls = 0.0f;
  for (int key = t; key < 1617; key += 256) {
    float e = __expf(pr[key] - M);
    pr[key] = e;
    ls += e;
  }
  red[t] = ls;
  __syncthreads();
  for (int ofs = 128; ofs; ofs >>= 1) {
    if (t < ofs) red[t] += red[t + ofs];
    __syncthreads();
  }
  float inv = 1.0f / red[0];
  if (t < 64) {
    float acc = 0.0f;
    for (int key = 0; key < 1617; ++key)
      acc += pr[key] * bf2f(base[(size_t)key * 2304 + 1536 + h * 64 + t]);
    attin[(size_t)b * 1617 * 768 + h * 64 + t] = f2bf(acc * inv);
  }
}

// ---------------- fused trajectory attention (WMMA) -------------------------
// block = (qtile, h, b); per frame f: logits (16x202) -> softmax -> attn@V
__global__ __launch_bounds__(256) void traj_attn_kernel(
    const u16* __restrict__ qkv, const u16* __restrict__ vt2,
    u16* __restrict__ traj, u16* __restrict__ xd) {
  __shared__ float lg[16 * 208];
  __shared__ u16 at[16 * 224];
  int qt = blockIdx.x, h = blockIdx.y, b = blockIdx.z;
  int lane = threadIdx.x & 31, wid = threadIdx.x >> 5;
  const u16* qbase = qkv + (size_t)b * 1617 * 2304;

  int qrow = 1 + qt * 16 + (lane & 15);
  if (qrow > 1616) qrow = 1616;

  for (int f = 0; f < 8; ++f) {
    // logits = q @ k^T  (both row-major in dh -> same fragment pattern)
    for (int nt = wid; nt < 13; nt += 8) {
      int krow = 1 + f * 202 + nt * 16 + (lane & 15);
      if (krow > 1616) krow = 1616;
      v8f acc = {};
      for (int kt = 0; kt < 2; ++kt) {
        Frag a, bk;
        load_frag_rm(a, qbase + h * 64, 2304, qrow, kt * 32, lane);
        load_frag_rm(bk, qbase + 768 + h * 64, 2304, krow, kt * 32, lane);
        acc = wmma_bf16(a, bk, acc);
      }
      for (int r = 0; r < 8; ++r)
        lg[(r + 8 * (lane >> 4)) * 208 + nt * 16 + (lane & 15)] = acc[r];
    }
    __syncthreads();

    // softmax over P=202, 16 threads per query, normalized bf16 into `at`
    {
      int q = threadIdx.x >> 4, i = threadIdx.x & 15;
      float mx = -1e30f;
      for (int c = i; c < 202; c += 16) mx = fmaxf(mx, lg[q * 208 + c]);
      for (int ofs = 8; ofs; ofs >>= 1)
        mx = fmaxf(mx, __shfl_xor(mx, ofs, 32));
      float sm = 0.0f;
      for (int c = i; c < 202; c += 16)
        sm += __expf((lg[q * 208 + c] - mx) * 0.125f);
      for (int ofs = 8; ofs; ofs >>= 1) sm += __shfl_xor(sm, ofs, 32);
      float inv = 1.0f / sm;
      for (int c = i; c < 224; c += 16) {
        float w = (c < 202)
                      ? __expf((lg[q * 208 + c] - mx) * 0.125f) * inv
                      : 0.0f;
        at[q * 224 + c] = f2bf(w);
      }
    }
    __syncthreads();

    // traj tile = attn (16x224, LDS) @ V (224x64, transposed layout)
    if (wid < 4) {
      int nt = wid;
      int d = nt * 16 + (lane & 15);
      const u16* vb =
          vt2 + ((((size_t)b * 12 + h) * 64 + d) * 8 + f) * 224;
      v8f acc = {};
      for (int kt = 0; kt < 7; ++kt) {
        Frag a, bv;
        load_frag_rm(a, at, 224, lane & 15, kt * 32, lane);
        int kg = (lane >> 4) * 8;
        bv.q[0] = *(const uint4*)(vb + kt * 32 + kg);
        bv.q[1] = *(const uint4*)(vb + kt * 32 + kg + 16);
        acc = wmma_bf16(a, bv, acc);
      }
      for (int r = 0; r < 8; ++r) {
        int sq = qt * 16 + r + 8 * (lane >> 4);
        if (sq < 1616) {
          u16 hv = f2bf(acc[r]);
          traj[(((size_t)b * 1616 + sq) * 8 + f) * 768 + h * 64 + d] = hv;
          if (f == sq / 202)
            xd[((size_t)b * 1616 + sq) * 768 + h * 64 + d] = hv;
        }
      }
    }
    __syncthreads();
  }
}

// ---------------- attn2 over F=8 + mixing (VALU) ----------------------------
__global__ __launch_bounds__(256) void attn2_mix_kernel(
    const u16* __restrict__ q2, const u16* __restrict__ k2,
    const u16* __restrict__ traj, u16* __restrict__ attin) {
  __shared__ float w[12][8];
  int bs = blockIdx.x;
  int b = bs / 1616, s = bs % 1616;
  int t = threadIdx.x;
  if (t < 96) {
    int h = t >> 3, f = t & 7;
    const u16* qp = q2 + (size_t)bs * 768 + h * 64;
    const u16* kp = k2 + ((size_t)bs * 8 + f) * 768 + h * 64;
    float dot = 0.0f;
    for (int d = 0; d < 64; ++d) dot += bf2f(qp[d]) * bf2f(kp[d]);
    w[h][f] = dot;
  }
  __syncthreads();
  if (t < 12) {
    float mx = -1e30f;
    for (int f = 0; f < 8; ++f) mx = fmaxf(mx, w[t][f]);
    float e[8], sm = 0.0f;
    for (int f = 0; f < 8; ++f) {
      e[f] = __expf(w[t][f] - mx);
      sm += e[f];
    }
    float inv = 1.0f / sm;
    for (int f = 0; f < 8; ++f) w[t][f] = e[f] * inv;
  }
  __syncthreads();
  for (int idx = t; idx < 768; idx += 256) {
    int h = idx >> 6;
    float acc = 0.0f;
    for (int f = 0; f < 8; ++f)
      acc += w[h][f] * bf2f(traj[((size_t)bs * 8 + f) * 768 + idx]);
    attin[((size_t)(b * 1617 + 1 + s)) * 768 + idx] = f2bf(acc);
  }
}

// ---------------- host side -------------------------------------------------
extern "C" void kernel_launch(void* const* d_in, const int* in_sizes, int n_in,
                              void* d_out, int out_size, void* d_ws,
                              size_t ws_size, hipStream_t stream) {
  (void)in_sizes; (void)n_in; (void)out_size; (void)ws_size;
  const float* x     = (const float*)d_in[0];
  const float* md    = (const float*)d_in[1];
  const float* Wqkv  = (const float*)d_in[2];
  const float* Wq2   = (const float*)d_in[3];
  const float* Wkv2  = (const float*)d_in[4];
  const float* Wproj = (const float*)d_in[5];
  const float* bproj = (const float*)d_in[6];
  const float* Wp1   = (const float*)d_in[7];
  const float* Wp2   = (const float*)d_in[8];
  const float* g1    = (const float*)d_in[9];
  const float* b1    = (const float*)d_in[10];
  float* out = (float*)d_out;

  char* w = (char*)d_ws;
  size_t off = 0;
  auto carve = [&](size_t bytes) -> void* {
    void* p = w + off;
    off += (bytes + 255) & ~(size_t)255;
    return p;
  };
  float* obj  = (float*)carve((size_t)96 * 768 * 4);
  u16* lnb    = (u16*)carve((size_t)3234 * 768 * 2);
  u16* qkvb   = (u16*)carve((size_t)3234 * 2304 * 2);
  u16* vt2    = (u16*)carve((size_t)24 * 64 * 8 * 224 * 2);
  u16* trajb  = (u16*)carve((size_t)25856 * 768 * 2);
  u16* xdb    = (u16*)carve((size_t)3232 * 768 * 2);
  u16* q2b    = (u16*)carve((size_t)3232 * 768 * 2);
  u16* k2b    = (u16*)carve((size_t)25856 * 768 * 2);
  u16* attin  = (u16*)carve((size_t)3234 * 768 * 2);
  u16* wqkvp  = (u16*)carve((size_t)768 * 2304 * 2);
  u16* wq2p   = (u16*)carve((size_t)768 * 768 * 2);
  u16* wkv2p  = (u16*)carve((size_t)768 * 768 * 2);
  u16* wprojp = (u16*)carve((size_t)768 * 768 * 2);
  u16* wp1p   = (u16*)carve((size_t)768 * 384 * 2);
  u16* wp2p   = (u16*)carve((size_t)384 * 768 * 2);

  zero_u16_kernel<<<2048, 256, 0, stream>>>(vt2, 24 * 64 * 8 * 224);

  pack_w<<<512, 256, 0, stream>>>(Wqkv,  wqkvp,  768, 2304, 2304);
  pack_w<<<512, 256, 0, stream>>>(Wq2,   wq2p,   768, 768,  768);
  pack_w<<<512, 256, 0, stream>>>(Wkv2,  wkv2p,  768, 768,  1536);  // k half
  pack_w<<<512, 256, 0, stream>>>(Wproj, wprojp, 768, 768,  768);
  pack_w<<<512, 256, 0, stream>>>(Wp1,   wp1p,   768, 384,  384);
  pack_w<<<512, 256, 0, stream>>>(Wp2,   wp2p,   384, 768,  768);

  roi_obj_kernel<<<96, 256, 0, stream>>>(x, md, wp1p, wp2p, obj);
  ln_kernel<<<3234, 256, 0, stream>>>(x, obj, g1, b1, lnb);

  // QKV projection (writes q,k row-major and v transposed)
  // rows=3234 -> 51 blocks of 64; N=2304 -> 9 blocks of 256
  gemm_wmma<<<dim3(51, 9), 256, 0, stream>>>(lnb, wqkvp, 3234, 768, 144, 0,
                                             1.0f, nullptr, nullptr, nullptr,
                                             nullptr, qkvb, vt2);

  cls_attn_kernel<<<24, 256, 0, stream>>>(qkvb, attin);
  traj_attn_kernel<<<dim3(101, 12, 2), 256, 0, stream>>>(qkvb, vt2, trajb,
                                                         xdb);

  // q2 = (xd @ Wq2) * scale   rows=3232 -> 51 blocks; N=768 -> 3 blocks
  gemm_wmma<<<dim3(51, 3), 256, 0, stream>>>(xdb, wq2p, 3232, 768, 48, 1,
                                             0.125f, q2b, nullptr, nullptr,
                                             nullptr, nullptr, nullptr);
  // k2 = traj @ Wkv2[:, :D]   rows=25856 -> 404 blocks exactly
  gemm_wmma<<<dim3(404, 3), 256, 0, stream>>>(trajb, wkv2p, 25856, 768, 48, 1,
                                              1.0f, k2b, nullptr, nullptr,
                                              nullptr, nullptr, nullptr);

  attn2_mix_kernel<<<3232, 256, 0, stream>>>(q2b, k2b, trajb, attin);

  // att @ Wproj + bproj + residual, with 202->196 token re-slice
  gemm_wmma<<<dim3(51, 3), 256, 0, stream>>>(attin, wprojp, 3234, 768, 48, 2,
                                             1.0f, nullptr, out, bproj, x,
                                             nullptr, nullptr);
}
